// GCBlockP1_70815420776691
// MI455X (gfx1250) — compile-verified, matrix operand order checked by reading, not run on hardware
//
#include <hip/hip_runtime.h>

// ---------------------------------------------------------------------------
// Types for CDNA5 WMMA (wave32, 16x16x32 bf16 -> f32)
// ---------------------------------------------------------------------------
typedef __attribute__((ext_vector_type(16))) __bf16 v16bf;
typedef __attribute__((ext_vector_type(8)))  __bf16 v8bf;
typedef __attribute__((ext_vector_type(8)))  float  v8f;
typedef __attribute__((ext_vector_type(4)))  float  v4f;
typedef __attribute__((ext_vector_type(4)))  int    v4i;

#define NDIM 128   // feature dim
#define NBAS 8     // basis dim

#if __has_builtin(__builtin_amdgcn_global_load_async_to_lds_b128)
#define HAVE_ASYNC_LDS 1
#else
#define HAVE_ASYNC_LDS 0
#endif

// Exact pointer types the async builtin expects: 16-byte int vector in
// global (AS1) / LDS (AS3) address spaces.
typedef __attribute__((address_space(1))) v4i g_v4i;
typedef __attribute__((address_space(3))) v4i l_v4i;

__device__ __forceinline__ v8f wmma_bf16(v16bf a, v16bf b, v8f c) {
    return __builtin_amdgcn_wmma_f32_16x16x32_bf16(
        /*neg_a=*/false, a, /*neg_b=*/false, b,
        /*c_mod=*/(short)0, c, /*reuse_a=*/false, /*reuse_b=*/false);
}

// CDNA5 hardware tanh (single trans-op instead of libm's branchy expansion).
__device__ __forceinline__ float fast_tanh(float x) {
#if __has_builtin(__builtin_amdgcn_tanhf)
    return __builtin_amdgcn_tanhf(x);
#elif __has_builtin(__builtin_amdgcn_tanh_f32)
    return __builtin_amdgcn_tanh_f32(x);
#else
    float r;
    asm("v_tanh_f32 %0, %1\n\tv_nop\n\tv_nop" : "=v"(r) : "v"(x));
    return r;
#endif
}

// Load a 16x32 bf16 operand fragment from a row-major [rows][ld] bf16 buffer.
// Per ISA 16-bit operand layout: lane l (l<16) holds row r0+l, K = {k0..k0+7,
// k0+16..k0+23}; lanes 16..31 hold the +8 K-half. Two b128 loads per lane.
__device__ __forceinline__ v16bf load_frag_bf16(const __bf16* p, int r0, int ld, int k0) {
    const int lane = threadIdx.x & 31;
    const int rr   = lane & 15;
    const int kh   = lane >> 4;
    const __bf16* q = p + (size_t)(r0 + rr) * ld + k0 + kh * 8;
    union { v16bf v; v8bf h[2]; } u;
    u.h[0] = *(const v8bf*)(q);
    u.h[1] = *(const v8bf*)(q + 16);
    return u.v;
}

// Same fragment shape, but source is f32 (LDS), converted to bf16 on the fly.
__device__ __forceinline__ v16bf load_frag_f32cvt(const float* p, int r0, int ld, int k0) {
    const int lane = threadIdx.x & 31;
    const int rr   = lane & 15;
    const int kh   = lane >> 4;
    const float* q = p + (size_t)(r0 + rr) * ld + k0 + kh * 8;
    v4f a0 = *(const v4f*)(q);
    v4f a1 = *(const v4f*)(q + 4);
    v4f a2 = *(const v4f*)(q + 16);
    v4f a3 = *(const v4f*)(q + 20);
    v16bf o;
    o[0]  = (__bf16)a0.x; o[1]  = (__bf16)a0.y; o[2]  = (__bf16)a0.z; o[3]  = (__bf16)a0.w;
    o[4]  = (__bf16)a1.x; o[5]  = (__bf16)a1.y; o[6]  = (__bf16)a1.z; o[7]  = (__bf16)a1.w;
    o[8]  = (__bf16)a2.x; o[9]  = (__bf16)a2.y; o[10] = (__bf16)a2.z; o[11] = (__bf16)a2.w;
    o[12] = (__bf16)a3.x; o[13] = (__bf16)a3.y; o[14] = (__bf16)a3.z; o[15] = (__bf16)a3.w;
    return o;
}

// ---------------------------------------------------------------------------
// Kernel 0: transpose + convert weights to bf16 K-major ("B-staged") layout.
// ---------------------------------------------------------------------------
__global__ __launch_bounds__(256) void prep_weights(
    const float* __restrict__ pp_w1, const float* __restrict__ pp_w2,
    const float* __restrict__ ii_w,  const float* __restrict__ pi_w,
    unsigned short* __restrict__ w1s_, unsigned short* __restrict__ w2s_,
    unsigned short* __restrict__ iis_, unsigned short* __restrict__ pis_) {
    __bf16* w1s = (__bf16*)w1s_;  __bf16* w2s = (__bf16*)w2s_;
    __bf16* iis = (__bf16*)iis_;  __bf16* pis = (__bf16*)pis_;
    int t = blockIdx.x * blockDim.x + threadIdx.x;
    if (t < NDIM * NDIM) {
        int n = t >> 7, k = t & (NDIM - 1);
        w1s[n * NDIM + k] = (__bf16)pp_w1[k * NDIM + n];
        w2s[n * NDIM + k] = (__bf16)pp_w2[k * NDIM + n];
        iis[n * NDIM + k] = (__bf16)ii_w[k * NDIM + n];
    }
    for (int i = t; i < 256 * 1024; i += gridDim.x * blockDim.x) {
        int n = i >> 8, k = i & 255;
        pis[i] = (__bf16)pi_w[(size_t)k * 1024 + n];
    }
}

// ---------------------------------------------------------------------------
// Kernel 1: node MLP  h = tanh(tanh(p1 W1 + b1) W2 + b2), output bf16.
// ---------------------------------------------------------------------------
__global__ __launch_bounds__(256) void pp_kernel(
    const float* __restrict__ p1,
    const float* __restrict__ b1, const float* __restrict__ b2,
    const unsigned short* __restrict__ w1s_, const unsigned short* __restrict__ w2s_,
    unsigned short* __restrict__ h_out_, int n_nodes) {
    const __bf16* w1s = (const __bf16*)w1s_;
    const __bf16* w2s = (const __bf16*)w2s_;
    __bf16* h_out = (__bf16*)h_out_;

    __shared__ __bf16 x_lds[64 * NDIM];   // 16 KB
    __shared__ __bf16 h_lds[64 * NDIM];   // 16 KB

    const int tid  = threadIdx.x;
    const int wave = tid >> 5;
    const int lane = tid & 31;
    const int kh   = lane >> 4;
    const int ln   = lane & 15;
    const int row0 = blockIdx.x * 64;

    for (int i = tid; i < 64 * NDIM / 4; i += 256) {
        int e4 = i * 4;
        int r = e4 >> 7, c = e4 & (NDIM - 1);
        int rg = min(row0 + r, n_nodes - 1);
        v4f v = *(const v4f*)(p1 + (size_t)rg * NDIM + c);
        x_lds[r * NDIM + c + 0] = (__bf16)v.x;
        x_lds[r * NDIM + c + 1] = (__bf16)v.y;
        x_lds[r * NDIM + c + 2] = (__bf16)v.z;
        x_lds[r * NDIM + c + 3] = (__bf16)v.w;
    }
    __syncthreads();

    const int nb = wave * 16;

    // ---- layer 1 ----
    {
        const float bias = b1[nb + ln];
        v16bf bfr[4];
        #pragma unroll
        for (int kt = 0; kt < 4; ++kt) bfr[kt] = load_frag_bf16(w1s, nb, NDIM, kt * 32);
        #pragma unroll
        for (int rt = 0; rt < 4; ++rt) {
            v8f acc;
            #pragma unroll
            for (int v = 0; v < 8; ++v) acc[v] = bias;
            v16bf a[4];
            #pragma unroll
            for (int kt = 0; kt < 4; ++kt) a[kt] = load_frag_bf16(x_lds, rt * 16, NDIM, kt * 32);
            #pragma unroll
            for (int kt = 0; kt < 4; ++kt) acc = wmma_bf16(a[kt], bfr[kt], acc);
            #pragma unroll
            for (int v = 0; v < 8; ++v) {
                int m = v + 8 * kh;
                h_lds[(rt * 16 + m) * NDIM + nb + ln] = (__bf16)fast_tanh(acc[v]);
            }
        }
    }
    __syncthreads();

    // ---- layer 2 ----
    {
        const float bias = b2[nb + ln];
        v16bf bfr[4];
        #pragma unroll
        for (int kt = 0; kt < 4; ++kt) bfr[kt] = load_frag_bf16(w2s, nb, NDIM, kt * 32);
        #pragma unroll
        for (int rt = 0; rt < 4; ++rt) {
            v8f acc;
            #pragma unroll
            for (int v = 0; v < 8; ++v) acc[v] = bias;
            v16bf a[4];
            #pragma unroll
            for (int kt = 0; kt < 4; ++kt) a[kt] = load_frag_bf16(h_lds, rt * 16, NDIM, kt * 32);
            #pragma unroll
            for (int kt = 0; kt < 4; ++kt) acc = wmma_bf16(a[kt], bfr[kt], acc);
            #pragma unroll
            for (int v = 0; v < 8; ++v) {
                int m  = v + 8 * kh;
                int rg = row0 + rt * 16 + m;
                if (rg < n_nodes)
                    h_out[(size_t)rg * NDIM + nb + ln] = (__bf16)fast_tanh(acc[v]);
            }
        }
    }
}

// ---------------------------------------------------------------------------
// Kernel 2: fused edge pipeline (see round-0 analysis for the math).
// ---------------------------------------------------------------------------
__global__ __launch_bounds__(256) void edge_kernel(
    const unsigned short* __restrict__ h_,
    const int* __restrict__ idx_i, const int* __restrict__ idx_j,
    const float* __restrict__ basis,
    const unsigned short* __restrict__ pis_, const float* __restrict__ pi_b,
    const unsigned short* __restrict__ iis_, const float* __restrict__ ii_b,
    float* __restrict__ out, int n_edges) {
    const __bf16* h   = (const __bf16*)h_;
    const __bf16* pis = (const __bf16*)pis_;
    const __bf16* iis = (const __bf16*)iis_;

    __shared__ __bf16 x_lds[64 * 256];      // 32 KB gathered inputs
    __shared__ float  inter_lds[64 * NDIM]; // 32 KB basis-contracted result
    __shared__ float  basis_lds[64 * NBAS]; // 2 KB
    __shared__ int    idxj_lds[64];

    const int tid  = threadIdx.x;
    const int wave = tid >> 5;
    const int lane = tid & 31;
    const int kh   = lane >> 4;
    const int ln   = lane & 15;
    const int e0   = blockIdx.x * 64;

    if (tid < 64) idxj_lds[tid] = idx_j[min(e0 + tid, n_edges - 1)];
    for (int i = tid; i < 64 * NBAS; i += 256) {
        int e = min(e0 + (i >> 3), n_edges - 1);
        basis_lds[i] = basis[(size_t)e * NBAS + (i & 7)];
    }
    // gather x rows: 2048 chunks of 8 bf16 (16 B each).
    // CDNA5 path: GLOBAL_LOAD_ASYNC_TO_LDS_B128 (per-lane global addr ->
    // per-lane LDS addr, tracked by ASYNCcnt, no VGPR staging).
    for (int ch = tid; ch < 64 * 32; ch += 256) {
        int r  = ch >> 5;
        int c8 = (ch & 31) * 8;
        int e  = min(e0 + r, n_edges - 1);
        int node = (c8 < NDIM) ? idx_i[e] : idx_j[e];
        const __bf16* src = h + (size_t)node * NDIM + (c8 & (NDIM - 1));
        __bf16*       dst = x_lds + r * 256 + c8;
#if HAVE_ASYNC_LDS
        // Integer round-trip sheds const + performs the required addrspace
        // retyping: AS1 int4* source, AS3 int4* (32-bit) LDS destination.
        __builtin_amdgcn_global_load_async_to_lds_b128(
            (g_v4i*)(size_t)src, (l_v4i*)(unsigned)(size_t)dst,
            /*offset=*/0, /*cpol=*/0);
#else
        *(v8bf*)dst = *(const v8bf*)src;
#endif
    }
#if HAVE_ASYNC_LDS
#if __has_builtin(__builtin_amdgcn_s_wait_asynccnt)
    __builtin_amdgcn_s_wait_asynccnt(0);
#else
    asm volatile("s_wait_asynccnt 0" ::: "memory");
#endif
#endif
    __syncthreads();

    // ---- pi GEMM + basis contraction ----
    const int colw = wave * 128;
    for (int ct = 0; ct < 8; ++ct) {
        const int   colz = colw + ct * 16 + ln;   // z-column for this lane
        const float pib  = pi_b[colz];
        __builtin_prefetch(pis + (size_t)colz * 256, 0, 1);  // global_prefetch
        v8f acc[4];
        #pragma unroll
        for (int rt = 0; rt < 4; ++rt)
            #pragma unroll
            for (int v = 0; v < 8; ++v) acc[rt][v] = 0.0f;
        for (int kt = 0; kt < 8; ++kt) {
            v16bf b = load_frag_bf16(pis, colw + ct * 16, 256, kt * 32);
            v16bf a[4];
            #pragma unroll
            for (int rt = 0; rt < 4; ++rt) a[rt] = load_frag_bf16(x_lds, rt * 16, 256, kt * 32);
            #pragma unroll
            for (int rt = 0; rt < 4; ++rt) acc[rt] = wmma_bf16(a[rt], b, acc[rt]);
        }
        const int chn  = colz >> 3;   // output channel (0..127)
        const int bidx = ln & 7;      // basis index
        #pragma unroll
        for (int rt = 0; rt < 4; ++rt) {
            #pragma unroll
            for (int v = 0; v < 8; ++v) {
                int   row = rt * 16 + v + 8 * kh;
                float val = (acc[rt][v] + pib) * basis_lds[row * NBAS + bidx];
                val += __shfl_xor(val, 1, 32);
                val += __shfl_xor(val, 2, 32);
                val += __shfl_xor(val, 4, 32);
                if (bidx == 0) inter_lds[row * NDIM + chn] = val;
            }
        }
    }
    __syncthreads();

    // ---- ii GEMM + tanh + scatter-add ----
    const int   nb   = wave * 16;
    const float bias = ii_b[nb + ln];
    v16bf bfr[4];
    #pragma unroll
    for (int kt = 0; kt < 4; ++kt) bfr[kt] = load_frag_bf16(iis, nb, NDIM, kt * 32);
    #pragma unroll
    for (int rt = 0; rt < 4; ++rt) {
        v8f acc;
        #pragma unroll
        for (int v = 0; v < 8; ++v) acc[v] = bias;
        v16bf a[4];
        #pragma unroll
        for (int kt = 0; kt < 4; ++kt) a[kt] = load_frag_f32cvt(inter_lds, rt * 16, NDIM, kt * 32);
        #pragma unroll
        for (int kt = 0; kt < 4; ++kt) acc = wmma_bf16(a[kt], bfr[kt], acc);
        #pragma unroll
        for (int v = 0; v < 8; ++v) {
            int row = rt * 16 + v + 8 * kh;
            if (e0 + row < n_edges) {
                int node = idxj_lds[row];
                atomicAdd(out + (size_t)node * NDIM + nb + ln, fast_tanh(acc[v]));
            }
        }
    }
}

// ---------------------------------------------------------------------------
// Host launcher
// ---------------------------------------------------------------------------
extern "C" void kernel_launch(void* const* d_in, const int* in_sizes, int n_in,
                              void* d_out, int out_size, void* d_ws, size_t ws_size,
                              hipStream_t stream) {
    const float* p1    = (const float*)d_in[0];
    const int*   idx_i = (const int*)d_in[1];
    const int*   idx_j = (const int*)d_in[2];
    const float* basis = (const float*)d_in[3];
    const float* pp_w1 = (const float*)d_in[4];
    const float* pp_b1 = (const float*)d_in[5];
    const float* pp_w2 = (const float*)d_in[6];
    const float* pp_b2 = (const float*)d_in[7];
    const float* pi_w  = (const float*)d_in[8];
    const float* pi_b  = (const float*)d_in[9];
    const float* ii_w  = (const float*)d_in[10];
    const float* ii_b  = (const float*)d_in[11];

    const int n_nodes = in_sizes[0] / NDIM;
    const int n_edges = in_sizes[1];

    char*  ws  = (char*)d_ws;
    size_t off = 0;
    unsigned short* h_bf = (unsigned short*)(ws + off); off += (size_t)n_nodes * NDIM * 2;
    off = (off + 255) & ~(size_t)255;
    unsigned short* w1s = (unsigned short*)(ws + off);  off += NDIM * NDIM * 2;
    unsigned short* w2s = (unsigned short*)(ws + off);  off += NDIM * NDIM * 2;
    unsigned short* iis = (unsigned short*)(ws + off);  off += NDIM * NDIM * 2;
    unsigned short* pis = (unsigned short*)(ws + off);  off += 256 * 1024 * 2;
    (void)ws_size; (void)n_in;

    (void)hipMemsetAsync(d_out, 0, (size_t)out_size * sizeof(float), stream);

    prep_weights<<<512, 256, 0, stream>>>(pp_w1, pp_w2, ii_w, pi_w, w1s, w2s, iis, pis);

    pp_kernel<<<(n_nodes + 63) / 64, 256, 0, stream>>>(
        p1, pp_b1, pp_b2, w1s, w2s, h_bf, n_nodes);

    edge_kernel<<<(n_edges + 63) / 64, 256, 0, stream>>>(
        h_bf, idx_i, idx_j, basis, pis, pi_b, iis, ii_b, (float*)d_out, n_edges);
}
